// AutoregressiveSelfAttention_83708912599078
// MI455X (gfx1250) — matrix operand order repaired and checked by reading, not compiled, not run on hardware
//
#include <hip/hip_runtime.h>
#include <hip/hip_bf16.h>
#include <math.h>

// ---------------------------------------------------------------------------
// Autoregressive self-attention for MI455X (gfx1250, wave32, WMMA bf16).
//   B=2, S=2048, H=2048, NH=16, HD=128.  out = Attn(RoPE(xWq^T), RoPE(xWk^T),
//   xWv^T) Wo^T  with scores scaled by sqrt(HD) (per reference) and causal mask.
// All GEMM loads use SADDR form: uniform base + one 32-bit lane offset, with
// row deltas folded into 24-bit instruction immediates (no 64-bit VGPR addrs).
// ---------------------------------------------------------------------------

typedef __bf16 bf16;
typedef __attribute__((ext_vector_type(16))) __bf16 v16bf;
typedef __attribute__((ext_vector_type(8)))  __bf16 v8bf;
typedef __attribute__((ext_vector_type(4)))  __bf16 v4bf;
typedef __attribute__((ext_vector_type(8)))  float  v8f;

#define B_   2
#define S_   2048
#define H_   2048
#define NH_  16
#define HD_  128
#define SCALE_ 11.313708498984761f   /* sqrt(128) — reference multiplies */

// ---- WMMA helper -----------------------------------------------------------
__device__ __forceinline__ v8f wmma_bf16(v16bf a, v16bf b, v8f c) {
  return __builtin_amdgcn_wmma_f32_16x16x32_bf16(
      /*neg_a=*/false, a, /*neg_b=*/false, b,
      /*c_mod=*/(short)0, c, /*reuse_a=*/false, /*reuse_b=*/false);
}

// A fragment (16x32 bf16).  `off` must already include kk + half*8 (elements);
// `rimm` is a compile-time row delta folded into the instruction offset.
//   elems 0..7 at off, elems 8..15 at off+16.
__device__ __forceinline__ v16bf a_bf16(const bf16* __restrict__ base,
                                        int off, int rimm) {
  const v8bf lo = *(const v8bf*)(base + off + rimm);
  const v8bf hi = *(const v8bf*)(base + off + rimm + 16);
  v16bf a;
#pragma unroll
  for (int e = 0; e < 8; ++e) { a[e] = lo[e]; a[8 + e] = hi[e]; }
  return a;
}

// A fragment from f32 source (convert to bf16 on the fly); same offset rules.
__device__ __forceinline__ v16bf a_f32(const float* __restrict__ base,
                                       int off, int rimm) {
  const float4 x0 = *(const float4*)(base + off + rimm);
  const float4 x1 = *(const float4*)(base + off + rimm + 4);
  const float4 x2 = *(const float4*)(base + off + rimm + 16);
  const float4 x3 = *(const float4*)(base + off + rimm + 20);
  v16bf a;
  a[0]  = (bf16)x0.x; a[1]  = (bf16)x0.y; a[2]  = (bf16)x0.z; a[3]  = (bf16)x0.w;
  a[4]  = (bf16)x1.x; a[5]  = (bf16)x1.y; a[6]  = (bf16)x1.z; a[7]  = (bf16)x1.w;
  a[8]  = (bf16)x2.x; a[9]  = (bf16)x2.y; a[10] = (bf16)x2.z; a[11] = (bf16)x2.w;
  a[12] = (bf16)x3.x; a[13] = (bf16)x3.y; a[14] = (bf16)x3.z; a[15] = (bf16)x3.w;
  return a;
}

// B fragment (32x16 bf16): `off` includes kk + half*8; elems 0..7 at off,
// elems 8..15 at off+16; `rimm` is a compile-time column-block delta.
__device__ __forceinline__ v16bf b_frag(const bf16* __restrict__ base,
                                        int off, int rimm) {
  const v8bf lo = *(const v8bf*)(base + off + rimm);
  const v8bf hi = *(const v8bf*)(base + off + rimm + 16);
  v16bf b;
#pragma unroll
  for (int e = 0; e < 8; ++e) { b[e] = lo[e]; b[8 + e] = hi[e]; }
  return b;
}

// ---------------------------------------------------------------------------
// Kernel 0: convert the four weight matrices f32 -> bf16 (vectorized x4).
// ---------------------------------------------------------------------------
__global__ void cvt_weights_kernel(const float* __restrict__ a,
                                   const float* __restrict__ b,
                                   const float* __restrict__ c,
                                   const float* __restrict__ d,
                                   bf16* __restrict__ oa, bf16* __restrict__ ob,
                                   bf16* __restrict__ oc, bf16* __restrict__ od) {
  size_t i = ((size_t)blockIdx.x * blockDim.x + threadIdx.x) * 4;  // < 4M*4
  float4 va = *(const float4*)(a + i);
  float4 vb = *(const float4*)(b + i);
  float4 vc = *(const float4*)(c + i);
  float4 vd = *(const float4*)(d + i);
  v4bf ta, tb, tc, td;
  ta[0] = (bf16)va.x; ta[1] = (bf16)va.y; ta[2] = (bf16)va.z; ta[3] = (bf16)va.w;
  tb[0] = (bf16)vb.x; tb[1] = (bf16)vb.y; tb[2] = (bf16)vb.z; tb[3] = (bf16)vb.w;
  tc[0] = (bf16)vc.x; tc[1] = (bf16)vc.y; tc[2] = (bf16)vc.z; tc[3] = (bf16)vc.w;
  td[0] = (bf16)vd.x; td[1] = (bf16)vd.y; td[2] = (bf16)vd.z; td[3] = (bf16)vd.w;
  *(v4bf*)(oa + i) = ta; *(v4bf*)(ob + i) = tb;
  *(v4bf*)(oc + i) = tc; *(v4bf*)(od + i) = td;
}

// ---------------------------------------------------------------------------
// Kernel 1: fused QKV projection + RoPE.
// One wave = 32 rows x one (d, d+64) column pair of one head.  Per 32-wide
// K step: 2 A frags + 6 B frags (all in distinct regs, loads clause ahead of
// the math) -> 12 WMMAs.  Only two 32-bit offset VGPRs carry addresses.
// ---------------------------------------------------------------------------
__global__ void __launch_bounds__(256, 1)
qkv_rope_kernel(const float* __restrict__ x,
                const bf16* __restrict__ Wq,
                const bf16* __restrict__ Wk,
                const bf16* __restrict__ Wv,
                const float* __restrict__ cosT,
                const float* __restrict__ sinT,
                bf16* __restrict__ qout,   // [B][NH][S][HD]
                bf16* __restrict__ kout,   // [B][NH][S][HD]
                bf16* __restrict__ vTout)  // [B][NH][HD][S]
{
  const int wave = threadIdx.x >> 5;
  const int lane = threadIdx.x & 31;
  const int l16 = lane & 15, half = lane >> 4;

  int task  = blockIdx.x * 8 + wave;   // 8192 tasks = 128 mblocks * 64 pairs
  int mBlk  = task >> 6;               // 32-row block
  int pair  = task & 63;
  int head  = pair >> 2;
  int d0    = (pair & 3) << 4;         // 0,16,32,48 (paired with +64)
  int mBase = mBlk << 5;

  // Per-lane 32-bit element offsets; row deltas go into instruction imms.
  int xoff = (mBase + l16) * H_ + half * 8;            // f32 elements
  int woff = (head * HD_ + d0 + l16) * H_ + half * 8;  // bf16 elements

  const v8f z = {0.f, 0.f, 0.f, 0.f, 0.f, 0.f, 0.f, 0.f};
  v8f cq0[2] = {z, z}, cq1[2] = {z, z};
  v8f ck0[2] = {z, z}, ck1[2] = {z, z};
  v8f cv0[2] = {z, z}, cv1[2] = {z, z};

  for (int kk = 0; kk < H_; kk += 32) {
    // stream-prefetch the next chunk of x (the only HBM-streamed operand)
    __builtin_prefetch(x + xoff + 64, 0, 0);
    __builtin_prefetch(x + xoff + 16 * H_ + 64, 0, 0);
    v16bf a0  = a_f32(x, xoff, 0);
    v16bf a1  = a_f32(x, xoff, 16 * H_);
    v16bf bq0 = b_frag(Wq, woff, 0);
    v16bf bq1 = b_frag(Wq, woff, 64 * H_);
    v16bf bk0 = b_frag(Wk, woff, 0);
    v16bf bk1 = b_frag(Wk, woff, 64 * H_);
    v16bf bv0 = b_frag(Wv, woff, 0);
    v16bf bv1 = b_frag(Wv, woff, 64 * H_);
    cq0[0] = wmma_bf16(a0, bq0, cq0[0]); cq0[1] = wmma_bf16(a1, bq0, cq0[1]);
    cq1[0] = wmma_bf16(a0, bq1, cq1[0]); cq1[1] = wmma_bf16(a1, bq1, cq1[1]);
    ck0[0] = wmma_bf16(a0, bk0, ck0[0]); ck0[1] = wmma_bf16(a1, bk0, ck0[1]);
    ck1[0] = wmma_bf16(a0, bk1, ck1[0]); ck1[1] = wmma_bf16(a1, bk1, ck1[1]);
    cv0[0] = wmma_bf16(a0, bv0, cv0[0]); cv0[1] = wmma_bf16(a1, bv0, cv0[1]);
    cv1[0] = wmma_bf16(a0, bv1, cv1[0]); cv1[1] = wmma_bf16(a1, bv1, cv1[1]);
    xoff += 32;
    woff += 32;
  }

  // RoPE on (d, d+64) pairs, then scatter to head-major layouts.
#pragma unroll
  for (int mi = 0; mi < 2; ++mi) {
#pragma unroll
    for (int v = 0; v < 8; ++v) {
      int mg = mBase + mi * 16 + v + 8 * half;   // global row in [0, B*S)
      int b  = mg >> 11;                         // /2048
      int s  = mg & 2047;
      int d  = d0 + l16;
      float c  = cosT[s * HD_ + d];              // cos[s][d] == cos[s][d+64]
      float sn = sinT[s * HD_ + d];
      float q0 = cq0[mi][v], q1 = cq1[mi][v];
      float k0 = ck0[mi][v], k1 = ck1[mi][v];
      size_t base = ((size_t)(b * NH_ + head) * S_ + s) * HD_;
      qout[base + d]      = (bf16)(q0 * c - q1 * sn);
      qout[base + d + 64] = (bf16)(q1 * c + q0 * sn);
      kout[base + d]      = (bf16)(k0 * c - k1 * sn);
      kout[base + d + 64] = (bf16)(k1 * c + k0 * sn);
      size_t vb = (size_t)(b * NH_ + head) * HD_;
      vTout[(vb + d)      * S_ + s] = (bf16)cv0[mi][v];
      vTout[(vb + d + 64) * S_ + s] = (bf16)cv1[mi][v];
    }
  }
}

// ---------------------------------------------------------------------------
// Kernel 2: causal flash attention.  One wave = one (b,h,16-row q tile).
// Per 32 k-positions: 8 score WMMAs + online softmax + P via LDS transpose +
// 8 P.V WMMAs.  k-loop bounded by the causal diagonal (half the work).
// ---------------------------------------------------------------------------
__global__ void __launch_bounds__(256, 1)
flash_attn_kernel(const bf16* __restrict__ q,
                  const bf16* __restrict__ k,
                  const bf16* __restrict__ vT,
                  bf16* __restrict__ ctx)  // [B][S][H]
{
  __shared__ bf16 lds[8][16 * 32];      // per-wave 16x32 P tile (8 KB total)
  const int wave = threadIdx.x >> 5;
  const int lane = threadIdx.x & 31;
  const int l16 = lane & 15, half = lane >> 4;

  int task  = blockIdx.x * 8 + wave;    // 4096 tasks = 32 (b*h) * 128 qtiles
  int qt    = task & 127;
  int bh    = task >> 7;                // b*16 + h
  int qBase = qt << 4;

  const bf16* qh = q  + (size_t)bh * S_ * HD_;   // uniform per wave (SGPRs)
  const bf16* kh = k  + (size_t)bh * S_ * HD_;
  const bf16* vh = vT + (size_t)bh * HD_ * S_;
  bf16* lp = lds[wave];

  // Per-lane constant element offsets (rows/kk folded into immediates).
  const int qoff = (qBase + l16) * HD_ + half * 8;
  const int koff = l16 * HD_ + half * 8;
  const int voff = l16 * S_ + half * 8;
  const int poff = l16 * 32 + half * 8;

  // Preload Q A-fragments for the whole k loop (16 x 128 tile -> 4 frags).
  v16bf qa[4];
#pragma unroll
  for (int i = 0; i < 4; ++i) qa[i] = a_bf16(qh, qoff + i * 32, 0);

  const v8f z = {0.f, 0.f, 0.f, 0.f, 0.f, 0.f, 0.f, 0.f};
  v8f o[8];
  float m[8], l[8];
#pragma unroll
  for (int i = 0; i < 8; ++i) { o[i] = z; m[i] = -3.0e38f; l[i] = 0.f; }

  const int kend = qBase + 16;          // exclusive causal bound for this tile
  for (int kb = 0; kb < kend; kb += 32) {
    const bf16* kbp = kh + (size_t)kb * HD_;   // uniform SGPR pointer advance
    const bf16* vbp = vh + kb;
    // --- scores: two 16-col tiles, K(hd)=128 in 4 WMMA steps each ---
    v8f s0 = z, s1 = z;
#pragma unroll
    for (int kk = 0; kk < HD_; kk += 32) {
      v16bf b0 = b_frag(kbp, koff, kk);
      v16bf b1 = b_frag(kbp, koff, kk + 16 * HD_);
      s0 = wmma_bf16(qa[kk >> 5], b0, s0);
      s1 = wmma_bf16(qa[kk >> 5], b1, s1);
    }
    // --- mask + online softmax (row lives across 16 lanes of one half) ---
#pragma unroll
    for (int v = 0; v < 8; ++v) {
      int qrow = qBase + v + 8 * half;
      float a0 = (kb + l16      <= qrow) ? s0[v] * SCALE_ : -3.0e38f;
      float a1 = (kb + 16 + l16 <= qrow) ? s1[v] * SCALE_ : -3.0e38f;
      float mx = fmaxf(a0, a1);
#pragma unroll
      for (int off = 1; off < 16; off <<= 1)
        mx = fmaxf(mx, __shfl_xor(mx, off, 32));
      float mn = fmaxf(m[v], mx);
      float p0 = __expf(a0 - mn), p1 = __expf(a1 - mn);
      float rs = p0 + p1;
#pragma unroll
      for (int off = 1; off < 16; off <<= 1)
        rs += __shfl_xor(rs, off, 32);
      float alpha = __expf(m[v] - mn);
      m[v] = mn;
      l[v] = l[v] * alpha + rs;
#pragma unroll
      for (int f = 0; f < 8; ++f) o[f][v] *= alpha;
      lp[(v + 8 * half) * 32 + l16]      = (bf16)p0;   // P -> LDS (transpose)
      lp[(v + 8 * half) * 32 + 16 + l16] = (bf16)p1;
    }
    // Wave-local LDS RAW: DS ops are in-order per wave; wait for data.
    asm volatile("s_wait_dscnt 0" ::: "memory");
    // --- ctx accumulate: P(16x32) x V(32x128) -> 8 WMMAs ---
    v16bf pa = a_bf16(lp, poff, 0);
#pragma unroll
    for (int f = 0; f < 8; ++f) {
      v16bf bv = b_frag(vbp, voff, f * 16 * S_);
      o[f] = wmma_bf16(pa, bv, o[f]);
    }
  }

  // Epilogue: 1/l scaling, write ctx row-major [B][S][H] for the final GEMM.
  int b = bh >> 4, h = bh & 15;
#pragma unroll
  for (int v = 0; v < 8; ++v) {
    float inv = 1.0f / l[v];
    int s = qBase + v + 8 * half;
    size_t row = ((size_t)b * S_ + s) * H_ + (size_t)h * HD_;
#pragma unroll
    for (int f = 0; f < 8; ++f)
      ctx[row + f * 16 + l16] = (bf16)(o[f][v] * inv);
  }
}

// ---------------------------------------------------------------------------
// Kernel 3: output projection  out = ctx @ Wo^T  (f32 result).
// One wave = 32x64 output tile (2M x 4N = 8 C frags); per 32-wide K step:
// 2 A frags + 4 B frags (distinct regs) -> 8 WMMAs.
// ---------------------------------------------------------------------------
__global__ void __launch_bounds__(256, 1)
out_proj_kernel(const bf16* __restrict__ ctx,
                const bf16* __restrict__ Wo,
                float* __restrict__ out) {
  const int wave = threadIdx.x >> 5;
  const int lane = threadIdx.x & 31;
  const int l16 = lane & 15, half = lane >> 4;

  int task  = blockIdx.x * 8 + wave;   // 4096 tasks = 128 mblocks * 32 nblocks
  int mBlk  = task >> 5;               // 32-row block
  int nBlk  = task & 31;               // 64-col block
  int mBase = mBlk << 5, n0 = nBlk << 6;

  int aoff = (mBase + l16) * H_ + half * 8;   // bf16 elements
  int boff = (n0 + l16) * H_ + half * 8;

  const v8f z = {0.f, 0.f, 0.f, 0.f, 0.f, 0.f, 0.f, 0.f};
  v8f c[2][4];
#pragma unroll
  for (int mi = 0; mi < 2; ++mi)
#pragma unroll
    for (int j = 0; j < 4; ++j) c[mi][j] = z;

  for (int kk = 0; kk < H_; kk += 32) {
    v16bf a0 = a_bf16(ctx, aoff, 0);
    v16bf a1 = a_bf16(ctx, aoff, 16 * H_);
    v16bf b0 = b_frag(Wo, boff, 0 * 16 * H_);
    v16bf b1 = b_frag(Wo, boff, 1 * 16 * H_);
    v16bf b2 = b_frag(Wo, boff, 2 * 16 * H_);
    v16bf b3 = b_frag(Wo, boff, 3 * 16 * H_);
    c[0][0] = wmma_bf16(a0, b0, c[0][0]); c[1][0] = wmma_bf16(a1, b0, c[1][0]);
    c[0][1] = wmma_bf16(a0, b1, c[0][1]); c[1][1] = wmma_bf16(a1, b1, c[1][1]);
    c[0][2] = wmma_bf16(a0, b2, c[0][2]); c[1][2] = wmma_bf16(a1, b2, c[1][2]);
    c[0][3] = wmma_bf16(a0, b3, c[0][3]); c[1][3] = wmma_bf16(a1, b3, c[1][3]);
    aoff += 32;
    boff += 32;
  }

#pragma unroll
  for (int mi = 0; mi < 2; ++mi)
#pragma unroll
    for (int v = 0; v < 8; ++v) {
      size_t row = (size_t)(mBase + mi * 16 + v + 8 * half) * H_;
#pragma unroll
      for (int j = 0; j < 4; ++j)
        out[row + n0 + j * 16 + l16] = c[mi][j][v];
    }
}

// ---------------------------------------------------------------------------
extern "C" void kernel_launch(void* const* d_in, const int* in_sizes, int n_in,
                              void* d_out, int out_size, void* d_ws,
                              size_t ws_size, hipStream_t stream) {
  (void)in_sizes; (void)n_in; (void)out_size; (void)ws_size;
  const float* x    = (const float*)d_in[0];
  const float* Wq   = (const float*)d_in[1];
  const float* Wk   = (const float*)d_in[2];
  const float* Wv   = (const float*)d_in[3];
  const float* Wo   = (const float*)d_in[4];
  const float* cosT = (const float*)d_in[5];
  const float* sinT = (const float*)d_in[6];
  float* out = (float*)d_out;

  // Workspace layout (bf16): 4 weights (4M each) + q,k,vT,ctx (8M each) = 96MB
  const size_t WN = (size_t)H_ * H_;         // 4M elements
  bf16* ws    = (bf16*)d_ws;
  bf16* Wq_b  = ws;
  bf16* Wk_b  = Wq_b + WN;
  bf16* Wv_b  = Wk_b + WN;
  bf16* Wo_b  = Wv_b + WN;
  bf16* q_b   = Wo_b + WN;                   // 2*WN elements each below
  bf16* k_b   = q_b + 2 * WN;
  bf16* vT_b  = k_b + 2 * WN;
  bf16* ctx_b = vT_b + 2 * WN;

  // 1) weights -> bf16 (4M elems / matrix, 4 elems/thread)
  cvt_weights_kernel<<<(int)(WN / 4 / 256), 256, 0, stream>>>(
      Wq, Wk, Wv, Wo, Wq_b, Wk_b, Wv_b, Wo_b);
  // 2) QKV + RoPE: 8192 wave-tasks / 8 waves per block
  qkv_rope_kernel<<<1024, 256, 0, stream>>>(x, Wq_b, Wk_b, Wv_b, cosT, sinT,
                                            q_b, k_b, vT_b);
  // 3) causal flash attention: 4096 wave-tasks / 8 waves per block
  flash_attn_kernel<<<512, 256, 0, stream>>>(q_b, k_b, vT_b, ctx_b);
  // 4) output projection: 4096 wave-tasks / 8 waves per block
  out_proj_kernel<<<512, 256, 0, stream>>>(ctx_b, Wo_b, out);
}
